// GConv_78340203479081
// MI455X (gfx1250) — matrix operand order repaired and checked by reading, not compiled, not run on hardware
//
#include <hip/hip_runtime.h>
#include <stdint.h>

// Problem constants (match reference)
#define D     128
#define SUMD  384
#define H     256
#define TM    128          // relations (rows) per workgroup
#define A_STRIDE 392       // bf16 elems per LDS A row (384 + 8 pad)
#define H_STRIDE 264       // bf16 elems per LDS h row (256 + 8 pad)

typedef __bf16 bf16;
typedef __attribute__((ext_vector_type(16))) bf16  v16bf;
typedef __attribute__((ext_vector_type(8)))  bf16  v8bf;
typedef __attribute__((ext_vector_type(4)))  bf16  v4bf;
typedef __attribute__((ext_vector_type(8)))  float v8f;
typedef __attribute__((ext_vector_type(4)))  float v4f;

static __device__ __forceinline__ bf16 f2bf(float f) {
  // round-to-nearest-even f32 -> bf16 without relying on fp conversion lowering
  unsigned u = __builtin_bit_cast(unsigned, f);
  unsigned r = (u + 0x7FFFu + ((u >> 16) & 1u)) >> 16;
  unsigned short s = (unsigned short)r;
  return __builtin_bit_cast(bf16, s);
}

// ---------------- prep kernels ----------------

__global__ void zero_acc_kernel(float* __restrict__ accF, float* __restrict__ cnt, int O) {
  long long idx = (long long)blockIdx.x * blockDim.x + threadIdx.x;
  long long tot = (long long)O * D;
  if (idx < tot) accF[idx] = 0.0f;
  if (idx < O)   cnt[idx]  = 0.0f;
}

// W1 is (SUMD x H) row-major [k][n]; produce W1T[n][k] bf16 (B-matrix friendly)
__global__ void cvt_w1_kernel(const float* __restrict__ W1, bf16* __restrict__ W1T) {
  int idx = blockIdx.x * blockDim.x + threadIdx.x;
  if (idx < SUMD * H) {
    int k = idx / H, n = idx % H;
    W1T[n * SUMD + k] = f2bf(W1[idx]);
  }
}

// W2 is (H x SUMD) row-major [k][n]; produce W2T[n][k] bf16
__global__ void cvt_w2_kernel(const float* __restrict__ W2, bf16* __restrict__ W2T) {
  int idx = blockIdx.x * blockDim.x + threadIdx.x;
  if (idx < H * SUMD) {
    int k = idx / SUMD, n = idx % SUMD;
    W2T[n * H + k] = f2bf(W2[idx]);
  }
}

__global__ void finalize_kernel(const float* __restrict__ accF, const float* __restrict__ cnt,
                                const float* __restrict__ objF, float* __restrict__ outF, int O) {
  long long idx = (long long)blockIdx.x * blockDim.x + threadIdx.x;
  if (idx < (long long)O * D) {
    float c = cnt[idx / D];
    c = (c < 1.0f) ? 1.0f : c;
    outF[idx] = accF[idx] / c + objF[idx];
  }
}

// ---------------- fused gather + MLP (WMMA) + scatter ----------------

__global__ __launch_bounds__(256, 1)
void gconv_mlp_kernel(const float* __restrict__ objF,
                      const float* __restrict__ relF,
                      const long long* __restrict__ trip,
                      const bf16* __restrict__ W1T, const float* __restrict__ b1,
                      const bf16* __restrict__ W2T, const float* __restrict__ b2,
                      float* __restrict__ accF, float* __restrict__ cnt,
                      float* __restrict__ relOut, int R)
{
  extern __shared__ bf16 smem[];   // A tile (128 x A_STRIDE); later aliased by h tile (128 x H_STRIDE)
  __shared__ int shSub[TM];
  __shared__ int shObj[TM];

  const int tid = threadIdx.x;
  const int blk = blockIdx.x;
  const int r0  = blk * TM;

  // ---- indices + degree counts ----
  if (tid < TM) {
    int r = r0 + tid;
    int s = 0, o = 0;
    if (r < R) {
      s = (int)trip[3ll * r + 0];
      o = (int)trip[3ll * r + 1];
      atomicAdd(&cnt[s], 1.0f);
      atomicAdd(&cnt[o], 1.0f);
    }
    shSub[tid] = s;
    shObj[tid] = o;
  }
  __syncthreads();

  // ---- gather [objF[sub] | relF[r] | objF[obj]] into LDS as bf16 ----
  {
    const int row  = tid >> 1;         // 2 threads per row
    const int half = tid & 1;          // 64 f32 columns each
    const int r    = r0 + row;
    const bool valid = (r < R);
    const float* seg0 = objF + (long long)shSub[row] * D;
    const float* seg1 = relF + (long long)r * D;
    const float* seg2 = objF + (long long)shObj[row] * D;
    const float* segs[3] = { seg0, seg1, seg2 };
    for (int sgi = 0; sgi < 3; ++sgi) {
      bf16* dst = smem + row * A_STRIDE + sgi * D + half * 64;
      if (valid) {
        const v4f* src = (const v4f*)(segs[sgi] + half * 64);
        #pragma unroll
        for (int j = 0; j < 16; ++j) {
          v4f f = src[j];
          v4bf o4;
          o4[0] = f2bf(f[0]); o4[1] = f2bf(f[1]);
          o4[2] = f2bf(f[2]); o4[3] = f2bf(f[3]);
          *(v4bf*)(dst + j * 4) = o4;
        }
      } else {
        v4bf z; z[0] = z[1] = z[2] = z[3] = f2bf(0.0f);
        #pragma unroll
        for (int j = 0; j < 16; ++j) *(v4bf*)(dst + j * 4) = z;
      }
    }
  }
  __syncthreads();

  const int wave  = tid >> 5;        // 8 waves
  const int lane  = tid & 31;
  const int strip = wave * 16;       // this wave's 16-row strip
  const int lrow  = lane & 15;       // A: M = lane%16 ; B/C/D: N = lane%16
  const int hi    = lane >> 4;       // K-half / M-half selector

  // ---- GEMM1: h = relu(A @ W1 + b1), h kept as bf16 fragments in registers ----
  v8bf hreg[H / 16];
  #pragma unroll
  for (int n = 0; n < H / 16; ++n) {
    float bias = b1[n * 16 + lrow];
    v8f acc;
    #pragma unroll
    for (int i = 0; i < 8; ++i) acc[i] = bias;
    #pragma unroll
    for (int k = 0; k < SUMD / 32; ++k) {
      // A fragment: lane holds row lrow,
      // 16 consecutive K at [k*32 + hi*16, +16) (matches B)
      const bf16* arow = smem + (strip + lrow) * A_STRIDE + k * 32 + hi * 16;
      v8bf a0 = *(const v8bf*)(arow);
      v8bf a1 = *(const v8bf*)(arow + 8);
      v16bf a = __builtin_shufflevector(a0, a1, 0,1,2,3,4,5,6,7,8,9,10,11,12,13,14,15);
      // B fragment: lane holds column N = n*16+lrow, 16 consecutive K at hi*16
      const v16bf* bp = (const v16bf*)(W1T + (long long)(n * 16 + lrow) * SUMD + k * 32 + hi * 16);
      v16bf b = *bp;
      acc = __builtin_amdgcn_wmma_f32_16x16x32_bf16(false, a, false, b,
                                                    (short)0, acc, false, false);
    }
    v8bf hb;
    #pragma unroll
    for (int i = 0; i < 8; ++i) {
      float v = acc[i] > 0.0f ? acc[i] : 0.0f;   // ReLU
      hb[i] = f2bf(v);
    }
    hreg[n] = hb;
  }
  __syncthreads();   // everyone finished reading the A tile

  // ---- spill h (C-layout frags) into LDS row-major, aliasing the A region ----
  bf16* hsm = smem;
  #pragma unroll
  for (int n = 0; n < H / 16; ++n) {
    #pragma unroll
    for (int i = 0; i < 8; ++i) {
      // C/D layout: VGPR i -> row i + 8*hi ; col = n*16 + lrow
      hsm[(strip + i + 8 * hi) * H_STRIDE + n * 16 + lrow] = hreg[n][i];
    }
  }
  __syncthreads();

  // per-lane row metadata for the epilogue (C-layout rows of this lane)
  int rowsub[8], rowobj[8], growg[8];
  bool rowok[8];
  #pragma unroll
  for (int i = 0; i < 8; ++i) {
    int rl = strip + i + 8 * hi;
    growg[i]  = r0 + rl;
    rowok[i]  = growg[i] < R;
    rowsub[i] = shSub[rl];
    rowobj[i] = shObj[rl];
  }

  // ---- GEMM2: out = h @ W2 + b2, fused split epilogue ----
  #pragma unroll
  for (int n = 0; n < SUMD / 16; ++n) {
    int col = n * 16 + lrow;
    float bias = b2[col];
    v8f acc;
    #pragma unroll
    for (int i = 0; i < 8; ++i) acc[i] = bias;
    #pragma unroll
    for (int k = 0; k < H / 32; ++k) {
      const bf16* arow = hsm + (strip + lrow) * H_STRIDE + k * 32 + hi * 16;
      v8bf a0 = *(const v8bf*)(arow);
      v8bf a1 = *(const v8bf*)(arow + 8);
      v16bf a = __builtin_shufflevector(a0, a1, 0,1,2,3,4,5,6,7,8,9,10,11,12,13,14,15);
      const v16bf* bp = (const v16bf*)(W2T + (long long)col * H + k * 32 + hi * 16);
      v16bf b = *bp;
      acc = __builtin_amdgcn_wmma_f32_16x16x32_bf16(false, a, false, b,
                                                    (short)0, acc, false, false);
    }
    if (n < 8) {                      // sub_out -> scatter-add
      int c = col;
      #pragma unroll
      for (int i = 0; i < 8; ++i)
        if (rowok[i]) atomicAdd(&accF[(long long)rowsub[i] * D + c], acc[i]);
    } else if (n < 16) {              // rel_out -> direct store
      int c = col - D;
      #pragma unroll
      for (int i = 0; i < 8; ++i)
        if (rowok[i]) relOut[(long long)growg[i] * D + c] = acc[i];
    } else {                          // obj_out -> scatter-add
      int c = col - 2 * D;
      #pragma unroll
      for (int i = 0; i < 8; ++i)
        if (rowok[i]) atomicAdd(&accF[(long long)rowobj[i] * D + c], acc[i]);
    }
  }
}

// ---------------- launcher ----------------

extern "C" void kernel_launch(void* const* d_in, const int* in_sizes, int n_in,
                              void* d_out, int out_size, void* d_ws, size_t ws_size,
                              hipStream_t stream) {
  const float*     objF = (const float*)d_in[0];
  const float*     relF = (const float*)d_in[1];
  const float*     W1   = (const float*)d_in[2];
  const float*     b1   = (const float*)d_in[3];
  const float*     W2   = (const float*)d_in[4];
  const float*     b2   = (const float*)d_in[5];
  const long long* trip = (const long long*)d_in[6];

  const int O = in_sizes[0] / D;
  const int R = in_sizes[1] / D;

  // workspace layout
  char*  ws   = (char*)d_ws;
  float* accF = (float*)ws;
  size_t off  = (size_t)O * D * sizeof(float);
  float* cnt  = (float*)(ws + off);
  off += (size_t)O * sizeof(float);
  off  = (off + 31) & ~(size_t)31;
  bf16* W1T = (bf16*)(ws + off);
  off += (size_t)H * SUMD * sizeof(bf16);
  off  = (off + 31) & ~(size_t)31;
  bf16* W2T = (bf16*)(ws + off);

  float* outFeat = (float*)d_out;
  float* relOut  = outFeat + (size_t)O * D;

  // prep (stream-ordered ahead of the fused kernel)
  {
    long long tot = (long long)O * D;
    int blocks = (int)((tot + 255) / 256);
    zero_acc_kernel<<<blocks, 256, 0, stream>>>(accF, cnt, O);
  }
  cvt_w1_kernel<<<(SUMD * H + 255) / 256, 256, 0, stream>>>(W1, W1T);
  cvt_w2_kernel<<<(H * SUMD + 255) / 256, 256, 0, stream>>>(W2, W2T);

  // fused gather + MLP + scatter
  {
    int nblk = (R + TM - 1) / TM;
    size_t lds = (size_t)TM * A_STRIDE * sizeof(bf16);   // ~98 KB (<< 320 KB WGP LDS)
    gconv_mlp_kernel<<<nblk, 256, lds, stream>>>(objF, relF, trip, W1T, b1, W2T, b2,
                                                 accF, cnt, relOut, R);
  }

  // output_feat = acc / max(cnt,1) + object_feats
  {
    long long tot = (long long)O * D;
    int blocks = (int)((tot + 255) / 256);
    finalize_kernel<<<blocks, 256, 0, stream>>>(accF, cnt, objF, outFeat, O);
  }

  // triplet + triplet_len pass-through (raw bytes after rel_out)
  char* tail = (char*)(relOut + (size_t)R * D);
  hipMemcpyAsync(tail, d_in[6], (size_t)R * 3 * sizeof(long long),
                 hipMemcpyDeviceToDevice, stream);
  hipMemcpyAsync(tail + (size_t)R * 3 * sizeof(long long), d_in[7], sizeof(int),
                 hipMemcpyDeviceToDevice, stream);
}
